// CapsNet_26688926777679
// MI455X (gfx1250) — matrix operand order repaired and checked by reading, not compiled, not run on hardware
//
#include <hip/hip_runtime.h>
#include <hip/hip_bf16.h>
#include <math.h>

// ---------------------------------------------------------------------------
// CapsNet forward on gfx1250 (MI455X). WMMA f16 GEMMs for the heavy convs and
// decoder; async global->LDS staging; f32 VALU for squash/routing/small ops.
// ---------------------------------------------------------------------------

typedef _Float16 v16h __attribute__((ext_vector_type(16)));
typedef float    v8f  __attribute__((ext_vector_type(8)));
typedef int      ai4  __attribute__((vector_size(16)));   // async-LDS builtin pointee

#define BATCH      256
#define NUM_ROUTES 1152
#define NUM_CAPS   10
#define PC_K       20736   // 256*81, = 648*32
#define PC_KSTEPS  648

#if __has_builtin(__builtin_amdgcn_global_load_async_to_lds_b128)
#define USE_ASYNC_LDS 1
#else
#define USE_ASYNC_LDS 0
#endif

// Copy 16 bytes global -> LDS (async on gfx1250 when available).
__device__ __forceinline__ void lds_cp16(_Float16* dst, const _Float16* src) {
#if USE_ASYNC_LDS
    __builtin_amdgcn_global_load_async_to_lds_b128(
        (__attribute__((address_space(1))) ai4*)src,
        (__attribute__((address_space(3))) ai4*)dst, 0, 0);
#else
    *(uint4*)dst = *(const uint4*)src;
#endif
}

__device__ __forceinline__ void stage_wait() {
#if USE_ASYNC_LDS
#if __has_builtin(__builtin_amdgcn_s_wait_asynccnt)
    __builtin_amdgcn_s_wait_asynccnt(0);
#else
    asm volatile("s_wait_asynccnt 0x0" ::: "memory");
#endif
#endif
}

// ---------------------------------------------------------------------------
// WMMA fragment loaders (LDS -> VGPR) following CDNA5 VGPR layouts.
// A tile: row-major [16 M][32 K] halves.  B tile: per-column [16 N][32 K].
// ---------------------------------------------------------------------------
__device__ __forceinline__ v16h load_frag_a(const _Float16* base) {
    int l  = threadIdx.x & 31;
    int m  = l & 15;
    int hi = (l >> 4) & 1;
    v16h f;
#pragma unroll
    for (int d = 0; d < 8; ++d) {
        int koff = ((d < 4) ? (d * 2) : (16 + (d - 4) * 2)) + hi * 8;
        f[2 * d]     = base[m * 32 + koff];
        f[2 * d + 1] = base[m * 32 + koff + 1];
    }
    return f;
}

__device__ __forceinline__ v16h load_frag_b(const _Float16* base) {
    int l  = threadIdx.x & 31;
    int n  = l & 15;
    int hi = (l >> 4) & 1;
    v16h f;
#pragma unroll
    for (int d = 0; d < 8; ++d) {
        int koff = ((d < 4) ? (d * 2) : (16 + (d - 4) * 2)) + hi * 8;
        f[2 * d]     = base[n * 32 + koff];
        f[2 * d + 1] = base[n * 32 + koff + 1];
    }
    return f;
}

__device__ __forceinline__ v8f wmma_f16(v16h a, v16h b, v8f c) {
    return __builtin_amdgcn_wmma_f32_16x16x32_f16(
        false, a, false, b, (short)0, c, false, false);
}

// ---------------------------------------------------------------------------
// Small utility kernels
// ---------------------------------------------------------------------------
__global__ void k_copy_f32(const float* __restrict__ src, float* __restrict__ dst, int n) {
    int i = blockIdx.x * blockDim.x + threadIdx.x;
    if (i < n) dst[i] = src[i];
}

__global__ void k_cvt_f16(const float* __restrict__ src, _Float16* __restrict__ dst, int n) {
    int i = blockIdx.x * blockDim.x + threadIdx.x;
    if (i < n) dst[i] = (_Float16)src[i];
}

// src [K][N] f32 -> dst [N][K] f16
__global__ void k_transpose_cvt(const float* __restrict__ src, _Float16* __restrict__ dst,
                                int K, int N) {
    int i = blockIdx.x * blockDim.x + threadIdx.x;
    if (i >= K * N) return;
    int k = i / N, n = i - k * N;
    dst[(size_t)n * K + k] = (_Float16)src[i];
}

__global__ void k_zero_f32(float* __restrict__ p, int n) {
    int i = blockIdx.x * blockDim.x + threadIdx.x;
    if (i < n) p[i] = 0.f;
}

// im2col offset LUT: ofs[k] = (k/81)*400 + ((k%81)/9)*20 + (k%81)%9
__global__ void k_make_ofs(int* __restrict__ ofs) {
    int k = blockIdx.x * blockDim.x + threadIdx.x;
    if (k >= PC_K) return;
    int c = k / 81, r = k - c * 81, kh = r / 9, kw = r - kh * 9;
    ofs[k] = c * 400 + kh * 20 + kw;
}

// ---------------------------------------------------------------------------
// conv1: 1->256, k9, s1, 28x28 -> 20x20, + bias + relu.  Direct conv, LDS tiled.
// grid = B*16 blocks, 256 thr; block = (b, 16 out-channels).
// ---------------------------------------------------------------------------
__global__ __launch_bounds__(256) void k_conv1(const float* __restrict__ x,
                                               const float* __restrict__ w,
                                               const float* __restrict__ bias,
                                               float* __restrict__ y) {
    __shared__ float sx[784];
    __shared__ float sw[16 * 81];
    int b   = blockIdx.x >> 4;
    int co0 = (blockIdx.x & 15) * 16;
    for (int i = threadIdx.x; i < 784; i += 256) sx[i] = x[(size_t)b * 784 + i];
    for (int i = threadIdx.x; i < 16 * 81; i += 256) sw[i] = w[(size_t)co0 * 81 + i];
    __syncthreads();
    int co = co0 + (threadIdx.x >> 4);
    const float* wp = &sw[(threadIdx.x >> 4) * 81];
    float bb = bias[co];
    for (int s = (threadIdx.x & 15); s < 400; s += 16) {
        int oh = s / 20, ow = s - oh * 20;
        float acc = bb;
#pragma unroll
        for (int kh = 0; kh < 9; ++kh)
#pragma unroll
            for (int kw = 0; kw < 9; ++kw)
                acc = fmaf(sx[(oh + kh) * 28 + ow + kw], wp[kh * 9 + kw], acc);
        y[((size_t)b * 256 + co) * 400 + s] = fmaxf(acc, 0.f);
    }
}

// ---------------------------------------------------------------------------
// PrimaryCaps implicit GEMM:  u[b][256][36] = Wf16[256][20736] x im2col(y[b]).
// Block = one image; 8 waves; wave w owns M rows [32w,32w+32) x 3 N-tiles.
// LDS double-buffered A panel (256x32 f16, async-staged) + im2col B panel.
// ---------------------------------------------------------------------------
__global__ __launch_bounds__(256) void k_primary_gemm(const _Float16* __restrict__ Aw,
                                                      const float* __restrict__ y,
                                                      const int* __restrict__ ofsLUT,
                                                      const float* __restrict__ pbias,
                                                      float* __restrict__ u) {
    __shared__ _Float16 sA[2][256 * 32];
    __shared__ _Float16 sB[2][48 * 32];
    int b    = blockIdx.x;
    int tid  = threadIdx.x;
    int wave = tid >> 5;
    const float* yb = y + (size_t)b * 256 * 400;

    v8f acc[2][3];
#pragma unroll
    for (int mi = 0; mi < 2; ++mi)
#pragma unroll
        for (int ni = 0; ni < 3; ++ni)
#pragma unroll
            for (int v = 0; v < 8; ++v) acc[mi][ni][v] = 0.f;

    // Per-thread im2col slot data (invariant across K-steps): 48*32/256 = 6 slots.
    int kks[6], sof[6];
#pragma unroll
    for (int j = 0; j < 6; ++j) {
        int idx = tid + j * 256;
        int n   = idx >> 5;
        kks[j]  = idx & 31;
        if (n < 36) {
            int oh = n / 6, ow = n - oh * 6;
            sof[j] = oh * 40 + ow * 2;   // (oh*2)*20 + (ow*2)
        } else {
            sof[j] = -1;                  // zero padding column
        }
    }

    auto stage = [&](int buf, int k0) {
        // A: thread t -> row m=t, 32 contiguous halves (64B), async global->LDS
        {
            const _Float16* src = Aw + (size_t)tid * PC_K + k0;
            _Float16* dst = &sA[buf][tid * 32];
            __builtin_prefetch(src + 32, 0, 1);
#pragma unroll
            for (int i = 0; i < 32; i += 8) lds_cp16(dst + i, src + i);
        }
        // B: im2col of y[b] via offset LUT, 6 slots/thread
#pragma unroll
        for (int j = 0; j < 6; ++j) {
            float val = 0.f;
            if (sof[j] >= 0) val = yb[ofsLUT[k0 + kks[j]] + sof[j]];
            sB[buf][tid + j * 256] = (_Float16)val;
        }
    };

    stage(0, 0);
    stage_wait();
    __syncthreads();
    for (int step = 0; step < PC_KSTEPS; ++step) {
        int buf = step & 1;
        if (step + 1 < PC_KSTEPS) stage(buf ^ 1, (step + 1) * 32);

        const _Float16* aBase = &sA[buf][(wave * 32) * 32];
        v16h a0 = load_frag_a(aBase);
        v16h a1 = load_frag_a(aBase + 16 * 32);
        v16h b0 = load_frag_b(&sB[buf][0]);
        v16h b1 = load_frag_b(&sB[buf][16 * 32]);
        v16h b2 = load_frag_b(&sB[buf][32 * 32]);

        acc[0][0] = wmma_f16(a0, b0, acc[0][0]);
        acc[0][1] = wmma_f16(a0, b1, acc[0][1]);
        acc[0][2] = wmma_f16(a0, b2, acc[0][2]);
        acc[1][0] = wmma_f16(a1, b0, acc[1][0]);
        acc[1][1] = wmma_f16(a1, b1, acc[1][1]);
        acc[1][2] = wmma_f16(a1, b2, acc[1][2]);

        stage_wait();
        __syncthreads();
    }

    int l    = tid & 31;
    int ncol = l & 15;
    int moff = (l >> 4) * 8;
#pragma unroll
    for (int mi = 0; mi < 2; ++mi)
#pragma unroll
        for (int ni = 0; ni < 3; ++ni)
#pragma unroll
            for (int v = 0; v < 8; ++v) {
                int m = wave * 32 + mi * 16 + moff + v;
                int n = ni * 16 + ncol;
                if (n < 36)
                    u[((size_t)b * 256 + m) * 36 + n] = acc[mi][ni][v] + pbias[m];
            }
}

// ---------------------------------------------------------------------------
// squash: u viewed [B][1152][8]; scale = sqrt(sn)/(1+sn)
// ---------------------------------------------------------------------------
__global__ void k_squash(const float* __restrict__ u, float* __restrict__ us, int total) {
    int i = blockIdx.x * blockDim.x + threadIdx.x;
    if (i >= total) return;
    const float* p = u + (size_t)i * 8;
    float vv[8], sn = 0.f;
#pragma unroll
    for (int j = 0; j < 8; ++j) { vv[j] = p[j]; sn = fmaf(vv[j], vv[j], sn); }
    float scale = sqrtf(sn) / (1.f + sn);
#pragma unroll
    for (int j = 0; j < 8; ++j) us[(size_t)i * 8 + j] = vv[j] * scale;
}

// ---------------------------------------------------------------------------
// u_hat[b,r,c,o] = sum_i W[r,c,o,i] * us[b,r,i]; thread per (b,r,c)
// ---------------------------------------------------------------------------
__global__ void k_uhat(const float* __restrict__ W, const float* __restrict__ us,
                       float* __restrict__ uh) {
    int idx = blockIdx.x * blockDim.x + threadIdx.x;
    if (idx >= BATCH * NUM_ROUTES * NUM_CAPS) return;
    int c  = idx % 10;
    int br = idx / 10;
    int r  = br % NUM_ROUTES;
    int b  = br / NUM_ROUTES;
    float ui[8];
    const float* up = us + ((size_t)b * NUM_ROUTES + r) * 8;
#pragma unroll
    for (int i = 0; i < 8; ++i) ui[i] = up[i];
    const float* wp = W + ((size_t)r * 10 + c) * 128;
    float* op = uh + (((size_t)b * NUM_ROUTES + r) * 10 + c) * 16;
#pragma unroll
    for (int o = 0; o < 16; ++o) {
        float a = 0.f;
#pragma unroll
        for (int i = 0; i < 8; ++i) a = fmaf(wp[o * 8 + i], ui[i], a);
        op[o] = a;
    }
}

// ---------------------------------------------------------------------------
// softmax of b_ij over routes (per caps column c). grid=10, 256 thr.
// ---------------------------------------------------------------------------
__global__ __launch_bounds__(256) void k_softmax_routes(const float* __restrict__ bij,
                                                        float* __restrict__ cij) {
    __shared__ float red[256];
    int c = blockIdx.x, t = threadIdx.x;
    float mx = -3.4e38f;
    for (int r = t; r < NUM_ROUTES; r += 256) mx = fmaxf(mx, bij[r * 10 + c]);
    red[t] = mx; __syncthreads();
    for (int s = 128; s > 0; s >>= 1) { if (t < s) red[t] = fmaxf(red[t], red[t + s]); __syncthreads(); }
    mx = red[0]; __syncthreads();
    float sum = 0.f;
    for (int r = t; r < NUM_ROUTES; r += 256) sum += expf(bij[r * 10 + c] - mx);
    red[t] = sum; __syncthreads();
    for (int s = 128; s > 0; s >>= 1) { if (t < s) red[t] += red[t + s]; __syncthreads(); }
    float inv = 1.f / red[0];
    for (int r = t; r < NUM_ROUTES; r += 256) cij[r * 10 + c] = expf(bij[r * 10 + c] - mx) * inv;
}

// ---------------------------------------------------------------------------
// s[b,c,o] = sum_r c[r,c]*uh[b,r,c,o];  v = s*sn/((1+sn)*sqrt(sn)) elementwise.
// grid = B*10, 128 thr (8 r-groups x 16 o).
// ---------------------------------------------------------------------------
__global__ __launch_bounds__(128) void k_s_v(const float* __restrict__ uh,
                                             const float* __restrict__ cij,
                                             float* __restrict__ v) {
    __shared__ float red[128];
    int b = blockIdx.x / 10, c = blockIdx.x % 10;
    int o = threadIdx.x & 15, g = threadIdx.x >> 4;
    float acc = 0.f;
    for (int r = g; r < NUM_ROUTES; r += 8)
        acc = fmaf(cij[r * 10 + c], uh[(((size_t)b * NUM_ROUTES + r) * 10 + c) * 16 + o], acc);
    red[threadIdx.x] = acc; __syncthreads();
    for (int s = 64; s >= 16; s >>= 1) {
        if (threadIdx.x < s) red[threadIdx.x] += red[threadIdx.x + s];
        __syncthreads();
    }
    if (threadIdx.x < 16) {
        float s  = red[threadIdx.x];
        float sn = s * s;
        float vv = sn * s / ((1.f + sn) * sqrtf(sn));
        v[((size_t)b * 10 + c) * 16 + threadIdx.x] = vv;
    }
}

// ---------------------------------------------------------------------------
// b_ij[r,c] += mean_b sum_o uh[b,r,c,o]*v[b,c,o]; thread per (r,c)
// ---------------------------------------------------------------------------
__global__ void k_agree(const float* __restrict__ uh, const float* __restrict__ v,
                        float* __restrict__ bij) {
    int idx = blockIdx.x * blockDim.x + threadIdx.x;
    if (idx >= NUM_ROUTES * NUM_CAPS) return;
    int c = idx % 10, r = idx / 10;
    float acc = 0.f;
    for (int b = 0; b < BATCH; ++b) {
        const float* up = uh + (((size_t)b * NUM_ROUTES + r) * 10 + c) * 16;
        const float* vp = v + ((size_t)b * 10 + c) * 16;
#pragma unroll
        for (int o = 0; o < 16; ++o) acc = fmaf(up[o], vp[o], acc);
    }
    bij[idx] += acc * (1.f / (float)BATCH);
}

// ---------------------------------------------------------------------------
// classes: cls[b,c] = ||v[b,c,:]||, softmax over BATCH axis (reference quirk).
// grid=10 (one c per block), 256 thr (one b per thread).
// ---------------------------------------------------------------------------
__global__ __launch_bounds__(256) void k_classes(const float* __restrict__ v,
                                                 float* __restrict__ cls_sm) {
    __shared__ float red[256];
    int c = blockIdx.x, b = threadIdx.x;
    const float* vp = v + ((size_t)b * 10 + c) * 16;
    float sn = 0.f;
#pragma unroll
    for (int o = 0; o < 16; ++o) sn = fmaf(vp[o], vp[o], sn);
    float cl = sqrtf(sn);
    red[b] = cl; __syncthreads();
    for (int s = 128; s > 0; s >>= 1) { if (b < s) red[b] = fmaxf(red[b], red[b + s]); __syncthreads(); }
    float mx = red[0]; __syncthreads();
    float e = expf(cl - mx);
    red[b] = e; __syncthreads();
    for (int s = 128; s > 0; s >>= 1) { if (b < s) red[b] += red[b + s]; __syncthreads(); }
    cls_sm[b * 10 + c] = e / red[0];
}

// ---------------------------------------------------------------------------
// argmax over caps, one-hot mask, write output/masked, build f16 rin[b][160]
// ---------------------------------------------------------------------------
__global__ void k_mask(const float* __restrict__ cls_sm, const float* __restrict__ v,
                       float* __restrict__ out_output, float* __restrict__ out_masked,
                       _Float16* __restrict__ rin) {
    int b = blockIdx.x * blockDim.x + threadIdx.x;
    if (b >= BATCH) return;
    int best = 0;
    float bv = cls_sm[b * 10 + 0];
    for (int c = 1; c < 10; ++c) {
        float t = cls_sm[b * 10 + c];
        if (t > bv) { bv = t; best = c; }
    }
    for (int c = 0; c < 10; ++c) {
        float mk = (c == best) ? 1.f : 0.f;
        out_masked[b * 10 + c] = mk;
        for (int o = 0; o < 16; ++o) {
            float val = v[((size_t)b * 10 + c) * 16 + o];
            out_output[((size_t)b * 10 + c) * 16 + o] = val;
            rin[(size_t)b * 160 + c * 16 + o] = (_Float16)(val * mk);
        }
    }
}

// ---------------------------------------------------------------------------
// Generic WMMA GEMM:  C[M,N] = act(A[M,K] @ Bt[N,K]^T + bias).
// act 0: relu, write f16 Aout.  act 1: sigmoid, write f32 Fout.
// Block tile 128M x 64N, 256 thr, 8 waves (4Mx2N), wave tile 32x32. K%32==0.
// ---------------------------------------------------------------------------
__global__ __launch_bounds__(256) void k_gemm_wmma(const _Float16* __restrict__ A,
                                                   const _Float16* __restrict__ Bt,
                                                   const float* __restrict__ bias,
                                                   int N, int K, int act,
                                                   _Float16* __restrict__ Aout,
                                                   float* __restrict__ Fout) {
    __shared__ _Float16 sA[2][128 * 32];
    __shared__ _Float16 sBt[2][64 * 32];
    int tid = threadIdx.x;
    int m0  = blockIdx.y * 128;
    int n0  = blockIdx.x * 64;
    int wave = tid >> 5;
    int wm = wave >> 1, wn = wave & 1;

    v8f acc[2][2];
#pragma unroll
    for (int mi = 0; mi < 2; ++mi)
#pragma unroll
        for (int ni = 0; ni < 2; ++ni)
#pragma unroll
            for (int v = 0; v < 8; ++v) acc[mi][ni][v] = 0.f;

    auto stage = [&](int buf, int k0) {
        {   // A: 128 rows x 32 halves; thread -> (row, 16-half chunk)
            int row = tid >> 1;
            int kc  = (tid & 1) * 16;
            const _Float16* src = A + (size_t)(m0 + row) * K + k0 + kc;
            __builtin_prefetch(src + 32, 0, 1);
            _Float16* dst = &sA[buf][row * 32 + kc];
            lds_cp16(dst, src);
            lds_cp16(dst + 8, src + 8);
        }
        {   // Bt: 64 rows x 32 halves; thread -> (n, 8-half chunk)
            int n  = tid >> 2;
            int kc = (tid & 3) * 8;
            int gn = n0 + n;
            _Float16* dst = &sBt[buf][n * 32 + kc];
            if (gn < N) {
                lds_cp16(dst, Bt + (size_t)gn * K + k0 + kc);
            } else {
#pragma unroll
                for (int i = 0; i < 8; ++i) dst[i] = (_Float16)0.f;
            }
        }
    };

    int steps = K >> 5;
    stage(0, 0);
    stage_wait();
    __syncthreads();
    for (int step = 0; step < steps; ++step) {
        int buf = step & 1;
        if (step + 1 < steps) stage(buf ^ 1, (step + 1) * 32);

        v16h a0 = load_frag_a(&sA[buf][(wm * 32) * 32]);
        v16h a1 = load_frag_a(&sA[buf][(wm * 32 + 16) * 32]);
        v16h b0 = load_frag_b(&sBt[buf][(wn * 32) * 32]);
        v16h b1 = load_frag_b(&sBt[buf][(wn * 32 + 16) * 32]);

        acc[0][0] = wmma_f16(a0, b0, acc[0][0]);
        acc[0][1] = wmma_f16(a0, b1, acc[0][1]);
        acc[1][0] = wmma_f16(a1, b0, acc[1][0]);
        acc[1][1] = wmma_f16(a1, b1, acc[1][1]);

        stage_wait();
        __syncthreads();
    }

    int l    = tid & 31;
    int ncol = l & 15;
    int moff = (l >> 4) * 8;
#pragma unroll
    for (int mi = 0; mi < 2; ++mi)
#pragma unroll
        for (int ni = 0; ni < 2; ++ni)
#pragma unroll
            for (int v = 0; v < 8; ++v) {
                int m = m0 + wm * 32 + mi * 16 + moff + v;
                int n = n0 + wn * 32 + ni * 16 + ncol;
                if (n < N) {
                    float val = acc[mi][ni][v] + bias[n];
                    if (act == 0) {
                        val = fmaxf(val, 0.f);
                        Aout[(size_t)m * N + n] = (_Float16)val;
                    } else {
                        val = 1.f / (1.f + expf(-val));
                        Fout[(size_t)m * N + n] = val;
                    }
                }
            }
}

// ---------------------------------------------------------------------------
// Host: workspace carving + launch sequence
// ---------------------------------------------------------------------------
static inline size_t align_up(size_t v, size_t a) { return (v + a - 1) & ~(a - 1); }

extern "C" void kernel_launch(void* const* d_in, const int* in_sizes, int n_in,
                              void* d_out, int out_size, void* d_ws, size_t ws_size,
                              hipStream_t stream) {
    (void)in_sizes; (void)n_in; (void)out_size; (void)ws_size;

    const float* x       = (const float*)d_in[0];
    const float* conv1_w = (const float*)d_in[1];
    const float* conv1_b = (const float*)d_in[2];
    const float* prim_w  = (const float*)d_in[3];
    const float* prim_b  = (const float*)d_in[4];
    const float* W_caps  = (const float*)d_in[5];
    const float* dec_w1  = (const float*)d_in[6];
    const float* dec_b1  = (const float*)d_in[7];
    const float* dec_w2  = (const float*)d_in[8];
    const float* dec_b2  = (const float*)d_in[9];
    const float* dec_w3  = (const float*)d_in[10];
    const float* dec_b3  = (const float*)d_in[11];

    float* out_x      = (float*)d_out;                       // [256,1,28,28]
    float* out_output = out_x + 200704;                      // [256,10,16,1]
    float* out_rec    = out_output + 40960;                  // [256,1,28,28]
    float* out_masked = out_rec + 200704;                    // [256,10]

    char* wsp = (char*)d_ws;
    size_t off = 0;
    auto carve = [&](size_t bytes) -> void* {
        void* p = wsp + off;
        off = align_up(off + bytes, 256);
        return p;
    };

    float*    y     = (float*)carve((size_t)BATCH * 256 * 400 * 4);   // conv1 out
    _Float16* Aprim = (_Float16*)carve((size_t)256 * PC_K * 2);       // prim_w f16
    int*      ofsL  = (int*)carve((size_t)PC_K * 4);                  // im2col LUT
    float*    u     = (float*)carve((size_t)BATCH * 9216 * 4);        // pre-squash
    float*    us    = (float*)carve((size_t)BATCH * 9216 * 4);        // squashed
    float*    uh    = (float*)carve((size_t)BATCH * NUM_ROUTES * 160 * 4); // u_hat
    float*    bij   = (float*)carve((size_t)NUM_ROUTES * 10 * 4);
    float*    cij   = (float*)carve((size_t)NUM_ROUTES * 10 * 4);
    float*    vbuf  = (float*)carve((size_t)BATCH * 160 * 4);
    float*    cls   = (float*)carve((size_t)BATCH * 10 * 4);
    _Float16* rin   = (_Float16*)carve((size_t)BATCH * 160 * 2);
    _Float16* w1t   = (_Float16*)carve((size_t)512 * 160 * 2);
    _Float16* w2t   = (_Float16*)carve((size_t)1024 * 512 * 2);
    _Float16* w3t   = (_Float16*)carve((size_t)784 * 1024 * 2);
    _Float16* h1    = (_Float16*)carve((size_t)BATCH * 512 * 2);
    _Float16* h2    = (_Float16*)carve((size_t)BATCH * 1024 * 2);

    const int T = 256;
    auto cdiv = [](int a, int b) { return (a + b - 1) / b; };

    // ---- prep: copy x to output, pack/transpose weights, LUT, zero b_ij ----
    k_copy_f32<<<cdiv(200704, T), T, 0, stream>>>(x, out_x, 200704);
    k_cvt_f16<<<cdiv(256 * PC_K, T), T, 0, stream>>>(prim_w, Aprim, 256 * PC_K);
    k_make_ofs<<<cdiv(PC_K, T), T, 0, stream>>>(ofsL);
    k_transpose_cvt<<<cdiv(160 * 512, T), T, 0, stream>>>(dec_w1, w1t, 160, 512);
    k_transpose_cvt<<<cdiv(512 * 1024, T), T, 0, stream>>>(dec_w2, w2t, 512, 1024);
    k_transpose_cvt<<<cdiv(1024 * 784, T), T, 0, stream>>>(dec_w3, w3t, 1024, 784);
    k_zero_f32<<<cdiv(NUM_ROUTES * 10, T), T, 0, stream>>>(bij, NUM_ROUTES * 10);

    // ---- conv1 + relu ----
    k_conv1<<<BATCH * 16, T, 0, stream>>>(x, conv1_w, conv1_b, y);

    // ---- PrimaryCaps implicit GEMM (WMMA) + bias ----
    k_primary_gemm<<<BATCH, T, 0, stream>>>(Aprim, y, ofsL, prim_b, u);

    // ---- squash ----
    k_squash<<<cdiv(BATCH * NUM_ROUTES, T), T, 0, stream>>>(u, us, BATCH * NUM_ROUTES);

    // ---- u_hat ----
    k_uhat<<<cdiv(BATCH * NUM_ROUTES * 10, T), T, 0, stream>>>(W_caps, us, uh);

    // ---- dynamic routing, 3 iterations ----
    for (int it = 0; it < 3; ++it) {
        k_softmax_routes<<<10, T, 0, stream>>>(bij, cij);
        k_s_v<<<BATCH * 10, 128, 0, stream>>>(uh, cij, vbuf);
        if (it < 2)
            k_agree<<<cdiv(NUM_ROUTES * 10, T), T, 0, stream>>>(uh, vbuf, bij);
    }

    // ---- classes (softmax over batch) + mask + rin ----
    k_classes<<<10, T, 0, stream>>>(vbuf, cls);
    k_mask<<<1, T, 0, stream>>>(cls, vbuf, out_output, out_masked, rin);

    // ---- decoder: 3 WMMA GEMMs ----
    k_gemm_wmma<<<dim3(512 / 64, 2), T, 0, stream>>>(rin, w1t, dec_b1, 512, 160, 0, h1, nullptr);
    k_gemm_wmma<<<dim3(1024 / 64, 2), T, 0, stream>>>(h1, w2t, dec_b2, 1024, 512, 0, h2, nullptr);
    k_gemm_wmma<<<dim3(cdiv(784, 64), 2), T, 0, stream>>>(h2, w3t, dec_b3, 784, 1024, 1, nullptr, out_rec);
}
